// MHA_1365799600947
// MI455X (gfx1250) — matrix operand order repaired and checked by reading, not compiled, not run on hardware
//
#include <hip/hip_runtime.h>

typedef _Float16 v16h __attribute__((ext_vector_type(16)));
typedef _Float16 v8h  __attribute__((ext_vector_type(8)));
typedef float    v8f  __attribute__((ext_vector_type(8)));

union F16x16 { v16h v; v8h h[2]; };

#define WMMA_F16(a, b, c) \
  __builtin_amdgcn_wmma_f32_16x16x32_f16(false, (a), false, (b), (short)0, (c), false, false)

#define NB   2
#define NT   2048
#define NC   1024
#define NH   16
#define DH   64

#define ZERO8 {0,0,0,0,0,0,0,0}

// ---------------------------------------------------------------------------
// fp32 -> fp16 conversion (grid-stride)
// ---------------------------------------------------------------------------
__global__ void cvt_f32_f16_kernel(const float* __restrict__ src,
                                   _Float16* __restrict__ dst, int n) {
  int i = blockIdx.x * blockDim.x + threadIdx.x;
  int stride = gridDim.x * blockDim.x;
  for (; i < n; i += stride) dst[i] = (_Float16)src[i];
}

// ---------------------------------------------------------------------------
// QKV projection: [4096,1024] x [1024,3072] -> Q,K (head-major), V (transposed)
// Wave computes a 32(M) x 64(N) tile: each B fragment reused for 2 A fragments.
// ---------------------------------------------------------------------------
__global__ __launch_bounds__(128) void gemm_qkv_kernel(
    const _Float16* __restrict__ Xh, const _Float16* __restrict__ Wh,
    _Float16* __restrict__ Qh, _Float16* __restrict__ Kh,
    _Float16* __restrict__ Vth) {
  const int lane = threadIdx.x & 31;
  const int wave = threadIdx.x >> 5;
  const int l16  = lane & 15;
  const int hl   = lane >> 4;                 // lane half (0/1)
  const int m0 = (blockIdx.x * 4 + wave) * 32;
  const int n0 = blockIdx.y * 64;

  const _Float16* Arow0 = Xh + (size_t)(m0 + l16) * NC;
  const _Float16* Arow1 = Xh + (size_t)(m0 + 16 + l16) * NC;

  v8f acc[2][4] = { { ZERO8, ZERO8, ZERO8, ZERO8 },
                    { ZERO8, ZERO8, ZERO8, ZERO8 } };

  for (int k = 0; k < NC; k += 32) {
    F16x16 a[2];                               // A: 16x32, lane-half K split 0-7/8-15
    const int kb = k + hl * 8;
    a[0].h[0] = *(const v8h*)(Arow0 + kb);
    a[0].h[1] = *(const v8h*)(Arow0 + kb + 16);
    a[1].h[0] = *(const v8h*)(Arow1 + kb);
    a[1].h[1] = *(const v8h*)(Arow1 + kb + 16);
    const int kb2 = k + hl * 8;                // B: lane-half K split 0-7/8-15 (+16)
#pragma unroll
    for (int c = 0; c < 4; ++c) {
      const _Float16* Brow = Wh + (size_t)(n0 + c * 16 + l16) * NC;
      F16x16 b;
      b.h[0] = *(const v8h*)(Brow + kb2);
      b.h[1] = *(const v8h*)(Brow + kb2 + 16);
      acc[0][c] = WMMA_F16(a[0].v, b.v, acc[0][c]);
      acc[1][c] = WMMA_F16(a[1].v, b.v, acc[1][c]);
    }
  }

#pragma unroll
  for (int s = 0; s < 2; ++s) {
#pragma unroll
    for (int c = 0; c < 4; ++c) {
      const int ncol = n0 + c * 16 + l16;
      const int seg = ncol >> 10;              // 0=Q 1=K 2=V
      const int cc = ncol & 1023;
      const int h = cc >> 6, d = cc & 63;
#pragma unroll
      for (int v = 0; v < 8; ++v) {
        const int row = m0 + s * 16 + v + hl * 8;   // C layout: M = v + 8*half
        const int bb = row >> 11, t = row & 2047;
        const float val = acc[s][c][v];
        if (seg == 0)
          Qh[((size_t)(bb * NH + h) * NT + t) * DH + d] = (_Float16)(val * 0.125f);
        else if (seg == 1)
          Kh[((size_t)(bb * NH + h) * NT + t) * DH + d] = (_Float16)val;
        else
          Vth[((size_t)(bb * NH + h) * DH + d) * NT + t] = (_Float16)val;
      }
    }
  }
}

// ---------------------------------------------------------------------------
// Flash attention. Block = 4 waves = 64 queries of one (b,h).
// The 32-key K/V block is staged in LDS once per block (4x global traffic cut);
// S^T = K_tile @ Q^T so the P fragment for P@V is built entirely in-lane.
// ---------------------------------------------------------------------------
__global__ __launch_bounds__(128) void flash_attn_kernel(
    const _Float16* __restrict__ Qh, const _Float16* __restrict__ Kh,
    const _Float16* __restrict__ Vth, _Float16* __restrict__ Yh) {
  __shared__ __align__(16) _Float16 sK[32 * 72];  // 32 keys x 64 d (stride 72: pad)
  __shared__ __align__(16) _Float16 sV[64 * 40];  // 64 d x 32 keys (stride 40: pad)

  const int tid  = threadIdx.x;
  const int lane = tid & 31;
  const int wave = tid >> 5;
  const int l16  = lane & 15;
  const int hl   = lane >> 4;
  const int qt = blockIdx.x * 4 + wave;        // 0..127
  const int bh = blockIdx.y;                   // 0..31
  const int q0 = qt * 16;

  const _Float16* Kbh = Kh + (size_t)bh * NT * DH;
  const _Float16* Vbh = Vth + (size_t)bh * DH * NT;

  // cooperative staging coordinates (each thread: 2 K chunks + 2 V chunks of 8 halves)
  const int kr = tid >> 3, kc = (tid & 7) * 8;  // K rows kr, kr+16
  const int vr = tid >> 2, vc = (tid & 3) * 8;  // V rows vr, vr+32

  // B-fragments of Q^T (reused for the whole key loop): lane = query column
  const _Float16* Qrow = Qh + ((size_t)bh * NT + q0 + l16) * DH;
  F16x16 bq0, bq1;
  {
    const int kb = hl * 8;
    bq0.h[0] = *(const v8h*)(Qrow + kb);
    bq0.h[1] = *(const v8h*)(Qrow + kb + 16);
    bq1.h[0] = *(const v8h*)(Qrow + 32 + kb);
    bq1.h[1] = *(const v8h*)(Qrow + 32 + kb + 16);
  }

  float m_st = -1e30f, l_st = 0.0f;
  v8f o[4] = { ZERO8, ZERO8, ZERO8, ZERO8 };

  for (int j = 0; j < NT; j += 32) {
    // ---- stage K (32x64) and V^T (64x32) blocks into LDS
    v8h kd0 = *(const v8h*)(Kbh + (size_t)(j + kr) * DH + kc);
    v8h kd1 = *(const v8h*)(Kbh + (size_t)(j + kr + 16) * DH + kc);
    v8h vd0 = *(const v8h*)(Vbh + (size_t)vr * NT + j + vc);
    v8h vd1 = *(const v8h*)(Vbh + (size_t)(vr + 32) * NT + j + vc);
    *(v8h*)(sK + (kr)      * 72 + kc) = kd0;
    *(v8h*)(sK + (kr + 16) * 72 + kc) = kd1;
    *(v8h*)(sV + (vr)      * 40 + vc) = vd0;
    *(v8h*)(sV + (vr + 32) * 40 + vc) = vd1;
    // speculative prefetch of the next key block
    __builtin_prefetch(Kbh + (size_t)(j + 32 + kr) * DH, 0, 0);
    __syncthreads();

    // ---- scores: S^T (keys x queries), two 16-key sub-tiles, K-dim = d = 64
    v8f st0 = ZERO8, st1 = ZERO8;
    {
      const _Float16* krow = sK + (l16) * 72;
      const int kb = hl * 8;
      F16x16 a0, a1;
      a0.h[0] = *(const v8h*)(krow + kb);
      a0.h[1] = *(const v8h*)(krow + kb + 16);
      a1.h[0] = *(const v8h*)(krow + 32 + kb);
      a1.h[1] = *(const v8h*)(krow + 32 + kb + 16);
      st0 = WMMA_F16(a0.v, bq0.v, st0);
      st0 = WMMA_F16(a1.v, bq1.v, st0);
    }
    {
      const _Float16* krow = sK + (16 + l16) * 72;
      const int kb = hl * 8;
      F16x16 a0, a1;
      a0.h[0] = *(const v8h*)(krow + kb);
      a0.h[1] = *(const v8h*)(krow + kb + 16);
      a1.h[0] = *(const v8h*)(krow + 32 + kb);
      a1.h[1] = *(const v8h*)(krow + 32 + kb + 16);
      st1 = WMMA_F16(a0.v, bq0.v, st1);
      st1 = WMMA_F16(a1.v, bq1.v, st1);
    }

    // ---- online softmax (lane owns query l16; halves hold keys {0-7,16-23}/{8-15,24-31})
    float mx = st0[0];
#pragma unroll
    for (int v = 1; v < 8; ++v) mx = fmaxf(mx, st0[v]);
#pragma unroll
    for (int v = 0; v < 8; ++v) mx = fmaxf(mx, st1[v]);
    mx = fmaxf(mx, __shfl_xor(mx, 16, 32));
    const float m_new = fmaxf(m_st, mx);
    const float alpha = __expf(m_st - m_new);

    float e0[8], e1[8], ls = 0.0f;
#pragma unroll
    for (int v = 0; v < 8; ++v) { e0[v] = __expf(st0[v] - m_new); ls += e0[v]; }
#pragma unroll
    for (int v = 0; v < 8; ++v) { e1[v] = __expf(st1[v] - m_new); ls += e1[v]; }
    ls += __shfl_xor(ls, 16, 32);
    l_st = l_st * alpha + ls;
    m_st = m_new;

    // ---- P as A-fragment (16 queries x 32 keys), built in-lane from S^T C-layout
    v16h pa;
#pragma unroll
    for (int v = 0; v < 8; ++v) { pa[v] = (_Float16)e0[v]; pa[8 + v] = (_Float16)e1[v]; }

    // ---- rescale O by alpha (per-row broadcast)
#pragma unroll
    for (int v = 0; v < 8; ++v) {
      const float av = __shfl(alpha, v + hl * 8, 32);
      o[0][v] *= av; o[1][v] *= av; o[2][v] *= av; o[3][v] *= av;
    }

    // ---- O += P @ V  (B-fragments from LDS-staged V^T: contiguous key runs)
#pragma unroll
    for (int c = 0; c < 4; ++c) {
      const _Float16* vrow = sV + (c * 16 + l16) * 40 + hl * 8;
      F16x16 bv;
      bv.h[0] = *(const v8h*)(vrow);
      bv.h[1] = *(const v8h*)(vrow + 16);
      o[c] = WMMA_F16(pa, bv.v, o[c]);
    }
    __syncthreads();
  }

  // ---- normalize and write y[b, t, h*64 + d] (f16, row-major for the out-proj GEMM)
  const int b = bh >> 4, h = bh & 15;
#pragma unroll
  for (int v = 0; v < 8; ++v) {
    const float lv = __shfl(l_st, v + hl * 8, 32);
    const float inv = 1.0f / lv;
    const int row = q0 + v + hl * 8;
    _Float16* yrow = Yh + ((size_t)(b * NT + row)) * NC + h * DH;
#pragma unroll
    for (int c = 0; c < 4; ++c)
      yrow[c * 16 + l16] = (_Float16)(o[c][v] * inv);
  }
}

// ---------------------------------------------------------------------------
// Output projection: [4096,1024] x [1024,1024] -> f32 d_out
// Wave computes 32(M) x 64(N).
// ---------------------------------------------------------------------------
__global__ __launch_bounds__(128) void gemm_out_kernel(
    const _Float16* __restrict__ Yh, const _Float16* __restrict__ Wh,
    float* __restrict__ out) {
  const int lane = threadIdx.x & 31;
  const int wave = threadIdx.x >> 5;
  const int l16  = lane & 15;
  const int hl   = lane >> 4;
  const int m0 = (blockIdx.x * 4 + wave) * 32;
  const int n0 = blockIdx.y * 64;

  const _Float16* Arow0 = Yh + (size_t)(m0 + l16) * NC;
  const _Float16* Arow1 = Yh + (size_t)(m0 + 16 + l16) * NC;

  v8f acc[2][4] = { { ZERO8, ZERO8, ZERO8, ZERO8 },
                    { ZERO8, ZERO8, ZERO8, ZERO8 } };

  for (int k = 0; k < NC; k += 32) {
    F16x16 a[2];
    const int kb = k + hl * 8;
    a[0].h[0] = *(const v8h*)(Arow0 + kb);
    a[0].h[1] = *(const v8h*)(Arow0 + kb + 16);
    a[1].h[0] = *(const v8h*)(Arow1 + kb);
    a[1].h[1] = *(const v8h*)(Arow1 + kb + 16);
    const int kb2 = k + hl * 8;
#pragma unroll
    for (int c = 0; c < 4; ++c) {
      const _Float16* Brow = Wh + (size_t)(n0 + c * 16 + l16) * NC;
      F16x16 b;
      b.h[0] = *(const v8h*)(Brow + kb2);
      b.h[1] = *(const v8h*)(Brow + kb2 + 16);
      acc[0][c] = WMMA_F16(a[0].v, b.v, acc[0][c]);
      acc[1][c] = WMMA_F16(a[1].v, b.v, acc[1][c]);
    }
  }

#pragma unroll
  for (int s = 0; s < 2; ++s) {
#pragma unroll
    for (int c = 0; c < 4; ++c) {
      const int col = n0 + c * 16 + l16;
#pragma unroll
      for (int v = 0; v < 8; ++v) {
        const int row = m0 + s * 16 + v + hl * 8;
        out[(size_t)row * NC + col] = acc[s][c][v];
      }
    }
  }
}

// ---------------------------------------------------------------------------
extern "C" void kernel_launch(void* const* d_in, const int* in_sizes, int n_in,
                              void* d_out, int out_size, void* d_ws, size_t ws_size,
                              hipStream_t stream) {
  (void)in_sizes; (void)n_in; (void)out_size; (void)ws_size;
  const float* x     = (const float*)d_in[0];
  const float* wqkv  = (const float*)d_in[1];
  const float* wproj = (const float*)d_in[2];

  char* ws = (char*)d_ws;
  _Float16* Xh     = (_Float16*)(ws);                        // 8 MiB (reused as Yh)
  _Float16* Wqkvh  = (_Float16*)(ws + ((size_t)8  << 20));   // 6 MiB
  _Float16* Wprojh = (_Float16*)(ws + ((size_t)14 << 20));   // 2 MiB
  _Float16* Qh     = (_Float16*)(ws + ((size_t)16 << 20));   // 8 MiB
  _Float16* Kh     = (_Float16*)(ws + ((size_t)24 << 20));   // 8 MiB
  _Float16* Vth    = (_Float16*)(ws + ((size_t)32 << 20));   // 8 MiB
  _Float16* Yh     = Xh;

  cvt_f32_f16_kernel<<<1024, 256, 0, stream>>>(x, Xh, NB * NT * NC);
  cvt_f32_f16_kernel<<<1024, 256, 0, stream>>>(wqkv, Wqkvh, 3 * NC * NC);
  cvt_f32_f16_kernel<<<512, 256, 0, stream>>>(wproj, Wprojh, NC * NC);

  // QKV GEMM: M=4096 (32 blocks x 4 waves x 32), N=3072 (48 x 64)
  gemm_qkv_kernel<<<dim3(32, 48), 128, 0, stream>>>(Xh, Wqkvh, Qh, Kh, Vth);

  // Flash attention: 128 query tiles (32 blocks x 4 waves) x 32 (b,h)
  flash_attn_kernel<<<dim3(32, 32), 128, 0, stream>>>(Qh, Kh, Vth, Yh);

  // Output projection: M=4096, N=1024 (16 x 64)
  gemm_out_kernel<<<dim3(32, 16), 128, 0, stream>>>(Yh, Wprojh, (float*)d_out);
}
